// DAEGCresci_87832081203317
// MI455X (gfx1250) — compile-verified, matrix operand order checked by reading, not run on hardware
//
#include <hip/hip_runtime.h>
#include <hip/hip_bf16.h>

// ---------------------------------------------------------------------------
// DAEGC suspicious-node scoring pipeline for MI455X (gfx1250, wave32, WMMA).
//
//   1) pack W1/W2/Wg1/Wg2 into f16 WMMA-B lane layout (one-time, tiny)
//   2) h1 = leaky(x @ W1 + b1)     WMMA f32_16x16x32_f16 (A fp32->f16 inline,
//   3) h2 = leaky(h1 @ W2 + b2)     B = two global_load_b128 of packed f16)
//   4) deg / rsqrt norm, conv1 + conv2 (WMMA weight mult, atomic aggregation)
//   5) logits/entropy + wave-reduced stats, edge cosine sim, final scores
//
// Output layout (700000 f32): logits[200k] | entropy | std_e | avg_sim | susp | mask
// ---------------------------------------------------------------------------

typedef __attribute__((ext_vector_type(16))) _Float16 v16h;
typedef __attribute__((ext_vector_type(8)))  float    v8f;

#define N_NODES 100000
#define N_EDGES 1600000
#define IN_DIM 256
#define HID 128
#define H2 64
#define NEG_SLOPE 0.01f

// ---- generic fill -----------------------------------------------------------
__global__ void k_fill(float* __restrict__ p, float v, int n) {
    int i = blockIdx.x * blockDim.x + threadIdx.x;
    int stride = gridDim.x * blockDim.x;
    for (; i < n; i += stride) p[i] = v;
}

// ---- pack fp32 weights [K,N] row-major into f16 WMMA B-fragment layout -----
// Tile (kt,nt): 512 halves; lane l owns halves [l*16 .. l*16+15] =
// W[kt*32 + (l>>4)*16 + i][nt*16 + (l&15)] for i in 0..15.
__global__ void k_pack_w(const float* __restrict__ W, _Float16* __restrict__ Wp,
                         int K, int N) {
    int idx = blockIdx.x * blockDim.x + threadIdx.x;
    int total = K * N;
    if (idx >= total) return;
    int tilesN = N >> 4;
    int tile = idx >> 9;           // 512 halves per 32x16 tile
    int r    = idx & 511;
    int lane = r >> 4;
    int i    = r & 15;
    int kt = tile / tilesN;
    int nt = tile - kt * tilesN;
    int k = kt * 32 + ((lane >> 4) << 4) + i;
    int n = nt * 16 + (lane & 15);
    Wp[idx] = (_Float16)W[(size_t)k * N + n];
}

// ---- WMMA GEMM: C[M,N] = act(A[M,K] @ W[K,N] + bias) -----------------------
// A: fp32, converted to f16 inline (read once). Wp: packed f16 fragments.
// One wave computes one 16x16 tile. M%16==0, N%16==0, K%32==0.
__global__ void k_gemm_wmma(const float* __restrict__ A,
                            const _Float16* __restrict__ Wp,
                            const float* __restrict__ bias,
                            float* __restrict__ C,
                            int M, int N, int K, int leaky) {
    int wave = (blockIdx.x * blockDim.x + threadIdx.x) >> 5;  // wave-uniform
    int lane = threadIdx.x & 31;
    int tilesN = N >> 4;
    int totalTiles = (M >> 4) * tilesN;
    if (wave >= totalTiles) return;                            // wave-uniform exit
    int tm = wave / tilesN;
    int tn = wave - tm * tilesN;

    int mA  = tm * 16 + (lane & 15);      // A-row for this lane
    int khi = (lane >> 4) << 4;           // K sub-block per half-wave

    const float* arow = A + (size_t)mA * K + khi;
    // this wave's packed B fragments: tile (kt, tn), lane-local 16 halves
    const _Float16* bbase = Wp + ((size_t)tn << 9) + ((size_t)lane << 4);
    size_t bstride = (size_t)tilesN << 9;  // halves per K-step across all n-tiles

    v8f c = {};
    #pragma unroll 2
    for (int k0 = 0; k0 < K; k0 += 32) {
        // A fragment: 16 contiguous fp32 -> 4x b128 loads + 8 cvt_pk
        const float4* ap = (const float4*)(arow + k0);
        float4 a0 = ap[0], a1 = ap[1], a2 = ap[2], a3 = ap[3];
        v16h a;
        a[0]  = (_Float16)a0.x; a[1]  = (_Float16)a0.y;
        a[2]  = (_Float16)a0.z; a[3]  = (_Float16)a0.w;
        a[4]  = (_Float16)a1.x; a[5]  = (_Float16)a1.y;
        a[6]  = (_Float16)a1.z; a[7]  = (_Float16)a1.w;
        a[8]  = (_Float16)a2.x; a[9]  = (_Float16)a2.y;
        a[10] = (_Float16)a2.z; a[11] = (_Float16)a2.w;
        a[12] = (_Float16)a3.x; a[13] = (_Float16)a3.y;
        a[14] = (_Float16)a3.z; a[15] = (_Float16)a3.w;
        // B fragment: 32 contiguous bytes of packed f16 -> 2x global_load_b128
        v16h b = *(const v16h*)(bbase + (size_t)(k0 >> 5) * bstride);
        c = __builtin_amdgcn_wmma_f32_16x16x32_f16(
                /*neg_a=*/false, a, /*neg_b=*/false, b,
                /*c_mod=*/(short)0, c, /*reuse_a=*/false, /*reuse_b=*/false);
    }

    int nIdx  = lane & 15;
    int mBase = tm * 16 + ((lane >> 4) << 3);
    float bv = bias ? bias[tn * 16 + nIdx] : 0.0f;
    #pragma unroll
    for (int r = 0; r < 8; ++r) {
        float v = c[r] + bv;
        if (leaky) v = (v >= 0.0f) ? v : NEG_SLOPE * v;
        C[(size_t)(mBase + r) * N + tn * 16 + nIdx] = v;
    }
}

// ---- degree over dst (self loop pre-filled with 1.0) -----------------------
__global__ void k_deg_edges(const long long* __restrict__ ei, float* __restrict__ deg, int E) {
    int e = blockIdx.x * blockDim.x + threadIdx.x;
    if (e < E) {
        int d = (int)ei[(size_t)E + e];
        atomicAdd(&deg[d], 1.0f);
    }
}

__global__ void k_invsqrt(const float* __restrict__ deg, float* __restrict__ invs, int n) {
    int i = blockIdx.x * blockDim.x + threadIdx.x;
    if (i < n) invs[i] = rsqrtf(fmaxf(deg[i], 1e-12f));
}

// ---- agg init: self-loop term agg[i] = hw[i] * invs[i]^2 -------------------
__global__ void k_agg_init(const float* __restrict__ hw, const float* __restrict__ invs,
                           float* __restrict__ agg, int n) {
    size_t i = (size_t)blockIdx.x * blockDim.x + threadIdx.x;
    size_t total = (size_t)n * H2;
    if (i < total) {
        float s = invs[i / H2];
        agg[i] = hw[i] * s * s;
    }
}

// ---- edge aggregation: wave per edge, lane owns 2 features -----------------
__global__ void k_agg_edges(const long long* __restrict__ ei,
                            const float* __restrict__ hw,
                            const float* __restrict__ invs,
                            float* __restrict__ agg, int E) {
    int gwave  = (blockIdx.x * blockDim.x + threadIdx.x) >> 5;
    int lane   = threadIdx.x & 31;
    int nwaves = (gridDim.x * blockDim.x) >> 5;
    for (int e = gwave; e < E; e += nwaves) {
        // sequential edge-index stream: speculative prefetch of next iteration
        __builtin_prefetch(ei + e + nwaves, 0, 1);
        __builtin_prefetch(ei + (size_t)E + e + nwaves, 0, 1);
        int s = (int)ei[e];
        int d = (int)ei[(size_t)E + e];
        float nrm = invs[s] * invs[d];
        float2 v = ((const float2*)(hw + (size_t)s * H2))[lane];
        atomicAdd(&agg[(size_t)d * H2 + 2 * lane + 0], v.x * nrm);
        atomicAdd(&agg[(size_t)d * H2 + 2 * lane + 1], v.y * nrm);
    }
}

// ---- finalize conv: out = leaky(agg + b) -----------------------------------
__global__ void k_conv_finish(const float* __restrict__ agg, const float* __restrict__ b,
                              float* __restrict__ out, int n) {
    size_t i = (size_t)blockIdx.x * blockDim.x + threadIdx.x;
    size_t total = (size_t)n * H2;
    if (i < total) {
        float v = agg[i] + b[i % H2];
        out[i] = (v >= 0.0f) ? v : NEG_SLOPE * v;
    }
}

// ---- logits + entropy + stats (sum, sumsq) ---------------------------------
__global__ void k_logits_entropy(const float* __restrict__ h,
                                 const float* __restrict__ Wc,
                                 const float* __restrict__ bc,
                                 float* __restrict__ out_logits,
                                 float* __restrict__ out_ent,
                                 float* __restrict__ stats, int n) {
    int i = blockIdx.x * blockDim.x + threadIdx.x;
    float ent = 0.0f;
    if (i < n) {
        const float4* hp = (const float4*)(h + (size_t)i * H2);
        float a0 = 0.0f, a1 = 0.0f;
        #pragma unroll
        for (int q = 0; q < H2 / 4; ++q) {
            float4 v = hp[q];
            int f = q * 4;
            a0 += v.x * Wc[(f + 0) * 2] + v.y * Wc[(f + 1) * 2] +
                  v.z * Wc[(f + 2) * 2] + v.w * Wc[(f + 3) * 2];
            a1 += v.x * Wc[(f + 0) * 2 + 1] + v.y * Wc[(f + 1) * 2 + 1] +
                  v.z * Wc[(f + 2) * 2 + 1] + v.w * Wc[(f + 3) * 2 + 1];
        }
        float l0 = a0 + bc[0], l1 = a1 + bc[1];
        float m  = fmaxf(l0, l1);
        float z  = __expf(l0 - m) + __expf(l1 - m);
        float lse = m + __logf(z);
        float lp0 = l0 - lse, lp1 = l1 - lse;
        float p0 = __expf(lp0), p1 = __expf(lp1);
        ent = -(p0 * lp0 + p1 * lp1);
        out_logits[(size_t)2 * i + 0] = l0;
        out_logits[(size_t)2 * i + 1] = l1;
        out_ent[i] = ent;
    }
    // wave-level reduction of sum & sumsq, one atomic pair per wave
    float se = ent, ss = ent * ent;
    #pragma unroll
    for (int o = 16; o > 0; o >>= 1) {
        se += __shfl_xor(se, o, 32);
        ss += __shfl_xor(ss, o, 32);
    }
    if ((threadIdx.x & 31) == 0) {
        atomicAdd(&stats[0], se);
        atomicAdd(&stats[1], ss);
    }
}

// ---- per-node inverse L2 norm ----------------------------------------------
__global__ void k_inv_norm(const float* __restrict__ h, float* __restrict__ invn, int n) {
    int i = blockIdx.x * blockDim.x + threadIdx.x;
    if (i < n) {
        const float4* hp = (const float4*)(h + (size_t)i * H2);
        float acc = 0.0f;
        #pragma unroll
        for (int q = 0; q < H2 / 4; ++q) {
            float4 v = hp[q];
            acc += v.x * v.x + v.y * v.y + v.z * v.z + v.w * v.w;
        }
        invn[i] = 1.0f / fmaxf(sqrtf(acc), 1e-8f);
    }
}

// ---- edge cosine sim: wave per edge, shfl_xor reduce, scatter atomics ------
__global__ void k_sim_edges(const long long* __restrict__ ei,
                            const float* __restrict__ h,
                            const float* __restrict__ invn,
                            float* __restrict__ sums,
                            float* __restrict__ cnt, int E) {
    int gwave  = (blockIdx.x * blockDim.x + threadIdx.x) >> 5;
    int lane   = threadIdx.x & 31;
    int nwaves = (gridDim.x * blockDim.x) >> 5;
    for (int e = gwave; e < E; e += nwaves) {
        __builtin_prefetch(ei + e + nwaves, 0, 1);
        __builtin_prefetch(ei + (size_t)E + e + nwaves, 0, 1);
        int s = (int)ei[e];
        int d = (int)ei[(size_t)E + e];
        float2 vs = ((const float2*)(h + (size_t)s * H2))[lane];
        float2 vd = ((const float2*)(h + (size_t)d * H2))[lane];
        float partial = vs.x * vd.x + vs.y * vd.y;
        #pragma unroll
        for (int o = 16; o > 0; o >>= 1) partial += __shfl_xor(partial, o, 32);
        if (lane == 0) {
            float sim = partial * invn[s] * invn[d];
            atomicAdd(&sums[s], sim);
            atomicAdd(&sums[d], sim);
            atomicAdd(&cnt[s], 1.0f);
            atomicAdd(&cnt[d], 1.0f);
        }
    }
}

// ---- final per-node score --------------------------------------------------
__global__ void k_final(const float* __restrict__ ent,
                        const float* __restrict__ sums,
                        const float* __restrict__ cnt,
                        const float* __restrict__ stats,
                        float* __restrict__ out_std,
                        float* __restrict__ out_avg,
                        float* __restrict__ out_susp,
                        float* __restrict__ out_mask, int n) {
    int i = blockIdx.x * blockDim.x + threadIdx.x;
    if (i >= n) return;
    float sum = stats[0], sumsq = stats[1];
    float fn = (float)n;
    float mean = sum / fn;
    float var = (sumsq - fn * mean * mean) / (fn - 1.0f);   // ddof=1
    float sd = sqrtf(fmaxf(var, 0.0f));
    float se = (ent[i] - mean) / (sd + 1e-8f);
    float cv = cnt[i];
    float as = (cv > 0.0f) ? (sums[i] / fmaxf(cv, 1.0f)) : 1.0f;
    float susp = 0.6f * se + 0.4f * (1.0f - as);
    out_std[i]  = se;
    out_avg[i]  = as;
    out_susp[i] = susp;
    out_mask[i] = ((susp > 1.0f) && (as < 0.35f)) ? 1.0f : 0.0f;
}

// ---- workspace float offsets -----------------------------------------------
#define OFF_H1    ((size_t)0)                       // 100000*128
#define OFF_HB    (OFF_H1   + (size_t)N_NODES*HID)  // 100000*64
#define OFF_HC    (OFF_HB   + (size_t)N_NODES*H2)
#define OFF_HD    (OFF_HC   + (size_t)N_NODES*H2)
#define OFF_DEG   (OFF_HD   + (size_t)N_NODES*H2)
#define OFF_INVS  (OFF_DEG  + (size_t)N_NODES)
#define OFF_INVN  (OFF_INVS + (size_t)N_NODES)
#define OFF_SUMS  (OFF_INVN + (size_t)N_NODES)
#define OFF_CNT   (OFF_SUMS + (size_t)N_NODES)
#define OFF_STATS (OFF_CNT  + (size_t)N_NODES)      // 16 floats
// packed f16 weights (sizes in floats; 2 halves per float); 32B-aligned
#define OFF_WP1   (OFF_STATS + (size_t)16)          // 256*128 halves = 16384 f
#define OFF_WP2   (OFF_WP1   + (size_t)16384)       // 128*64 halves  = 4096 f
#define OFF_WPG1  (OFF_WP2   + (size_t)4096)        // 64*64 halves   = 2048 f
#define OFF_WPG2  (OFF_WPG1  + (size_t)2048)        // 64*64 halves   = 2048 f

extern "C" void kernel_launch(void* const* d_in, const int* in_sizes, int n_in,
                              void* d_out, int out_size, void* d_ws, size_t ws_size,
                              hipStream_t stream) {
    (void)in_sizes; (void)n_in; (void)out_size; (void)ws_size;

    const float*     x   = (const float*)d_in[0];
    const long long* ei  = (const long long*)d_in[1];   // int64 edge_index [2,E]
    const float*     W1  = (const float*)d_in[2];
    const float*     b1  = (const float*)d_in[3];
    const float*     W2  = (const float*)d_in[4];
    const float*     b2  = (const float*)d_in[5];
    const float*     Wg1 = (const float*)d_in[6];
    const float*     bg1 = (const float*)d_in[7];
    const float*     Wg2 = (const float*)d_in[8];
    const float*     bg2 = (const float*)d_in[9];
    const float*     Wc  = (const float*)d_in[10];
    const float*     bc  = (const float*)d_in[11];

    float* ws    = (float*)d_ws;
    float* h1    = ws + OFF_H1;
    float* hB    = ws + OFF_HB;
    float* hC    = ws + OFF_HC;
    float* hD    = ws + OFF_HD;
    float* deg   = ws + OFF_DEG;
    float* invs  = ws + OFF_INVS;
    float* invn  = ws + OFF_INVN;
    float* sums  = ws + OFF_SUMS;
    float* cnt   = ws + OFF_CNT;
    float* stats = ws + OFF_STATS;
    _Float16* Wp1  = (_Float16*)(ws + OFF_WP1);
    _Float16* Wp2  = (_Float16*)(ws + OFF_WP2);
    _Float16* Wpg1 = (_Float16*)(ws + OFF_WPG1);
    _Float16* Wpg2 = (_Float16*)(ws + OFF_WPG2);

    float* out        = (float*)d_out;
    float* out_logits = out;                 // 200000
    float* out_ent    = out + 200000;        // 100000
    float* out_std    = out + 300000;
    float* out_avg    = out + 400000;
    float* out_susp   = out + 500000;
    float* out_mask   = out + 600000;

    const int TB = 256;
    auto gemm = [&](const float* A, const _Float16* Wp, const float* bias, float* C,
                    int M, int N, int K, int leaky) {
        int tiles = (M >> 4) * (N >> 4);
        int blocks = (tiles * 32 + TB - 1) / TB;
        k_gemm_wmma<<<blocks, TB, 0, stream>>>(A, Wp, bias, C, M, N, K, leaky);
    };

    int nodeBlk  = (N_NODES + TB - 1) / TB;
    int featBlk  = ((N_NODES * H2) + TB - 1) / TB;
    int edgeBlk  = (N_EDGES + TB - 1) / TB;
    int edgeWaveBlk = 4096;   // grid-stride, 32768 waves over 1.6M edges

    // 0) pack weights into f16 WMMA B-fragment layout (tiny)
    k_pack_w<<<(IN_DIM * HID + TB - 1) / TB, TB, 0, stream>>>(W1,  Wp1,  IN_DIM, HID);
    k_pack_w<<<(HID * H2 + TB - 1) / TB,     TB, 0, stream>>>(W2,  Wp2,  HID,    H2);
    k_pack_w<<<(H2 * H2 + TB - 1) / TB,      TB, 0, stream>>>(Wg1, Wpg1, H2,     H2);
    k_pack_w<<<(H2 * H2 + TB - 1) / TB,      TB, 0, stream>>>(Wg2, Wpg2, H2,     H2);

    // 1) encoder
    gemm(x,  Wp1, b1, h1, N_NODES, HID, IN_DIM, 1);
    gemm(h1, Wp2, b2, hB, N_NODES, H2,  HID,    1);

    // 2) degrees (self loop = 1) and sym-norm
    k_fill<<<nodeBlk, TB, 0, stream>>>(deg, 1.0f, N_NODES);
    k_deg_edges<<<edgeBlk, TB, 0, stream>>>(ei, deg, N_EDGES);
    k_invsqrt<<<nodeBlk, TB, 0, stream>>>(deg, invs, N_NODES);

    // 3) GCN conv 1: hB -> (leaky agg) -> h1 (reused as 64-dim buffer)
    gemm(hB, Wpg1, nullptr, hC, N_NODES, H2, H2, 0);
    k_agg_init<<<featBlk, TB, 0, stream>>>(hC, invs, hD, N_NODES);
    k_agg_edges<<<edgeWaveBlk, TB, 0, stream>>>(ei, hC, invs, hD, N_EDGES);
    k_conv_finish<<<featBlk, TB, 0, stream>>>(hD, bg1, h1, N_NODES);

    // 4) GCN conv 2: h1 -> hB (final node embedding h)
    gemm(h1, Wpg2, nullptr, hC, N_NODES, H2, H2, 0);
    k_agg_init<<<featBlk, TB, 0, stream>>>(hC, invs, hD, N_NODES);
    k_agg_edges<<<edgeWaveBlk, TB, 0, stream>>>(ei, hC, invs, hD, N_EDGES);
    k_conv_finish<<<featBlk, TB, 0, stream>>>(hD, bg2, hB, N_NODES);

    // 5) logits + entropy + entropy stats
    k_fill<<<1, 64, 0, stream>>>(stats, 0.0f, 2);
    k_logits_entropy<<<nodeBlk, TB, 0, stream>>>(hB, Wc, bc, out_logits, out_ent,
                                                 stats, N_NODES);

    // 6) cosine similarity over edges
    k_inv_norm<<<nodeBlk, TB, 0, stream>>>(hB, invn, N_NODES);
    k_fill<<<nodeBlk, TB, 0, stream>>>(sums, 0.0f, N_NODES);
    k_fill<<<nodeBlk, TB, 0, stream>>>(cnt, 0.0f, N_NODES);
    k_sim_edges<<<edgeWaveBlk, TB, 0, stream>>>(ei, hB, invn, sums, cnt, N_EDGES);

    // 7) final scores
    k_final<<<nodeBlk, TB, 0, stream>>>(out_ent, sums, cnt, stats,
                                        out_std, out_avg, out_susp, out_mask, N_NODES);
}
